// GeometryAttention_15951508537766
// MI455X (gfx1250) — matrix operand order repaired and checked by reading, not compiled, not run on hardware
//
#include <hip/hip_runtime.h>

// ---------------------------------------------------------------------------
// GeometryAttention on gfx1250 (MI455X): all GEMM-shaped work via
// v_wmma_f32_16x16x32_f16 (f16 A/B, f32 accumulate), wave32, WGP-resident LDS
// tiles. Problem sizes: B=8, N=1024, F=512, G=128, H=8, D=64.
// ---------------------------------------------------------------------------

typedef _Float16 v16h __attribute__((ext_vector_type(16)));
typedef _Float16 v8h  __attribute__((ext_vector_type(8)));
typedef float    v8f  __attribute__((ext_vector_type(8)));

union HFrag { v16h v; v8h h[2]; _Float16 e[16]; };
union FFrag { v8f v; float f[8]; };

static __device__ __forceinline__ v8f wmma_f16(const HFrag& a, const HFrag& b, v8f c) {
  // D = A(16x32 f16) x B(32x16 f16) + C(16x16 f32)
  return __builtin_amdgcn_wmma_f32_16x16x32_f16(false, a.v, false, b.v,
                                                (short)0, c, false, false);
}

static __device__ __forceinline__ float gelu_tanh(float x) {
  float x3 = x * x * x;
  return 0.5f * x * (1.0f + tanhf(0.7978845608028654f * (x + 0.044715f * x3)));
}

constexpr int Bc = 8, Nn = 1024, Fc = 512, Gc = 128, Hc = 8, Dc = 64;

// ---------------------------------------------------------------------------
// Generic WMMA GEMM: C[b] = A[b] (MxK) * Bw[b] (KxN) (+ bias) (+= if ACCUM)
// Requires M%64==0, K%32==0, N%64==0 (true for every call site here).
// Block: 128 threads = 4 waves; each wave computes a 32x32 sub-tile of the
// 64x64 block tile. fp32 global -> f16 LDS conversion on the fly.
// HAS_BIAS/ACCUM are compile-time so the epilogue is branch-free.
// ---------------------------------------------------------------------------
template <bool HAS_BIAS, bool ACCUM>
__global__ __launch_bounds__(128) void gemm_wmma_kernel(
    const float* __restrict__ A, const float* __restrict__ Bw,
    const float* __restrict__ bias, float* __restrict__ C,
    int M, int K, int N,
    long sA, long sB, long sC)
{
  constexpr int LDA = 40;   // padded halves/row: 80B, 16B-aligned strides
  __shared__ __align__(16) _Float16 As[64 * LDA];   // A tile  [m][k]
  __shared__ __align__(16) _Float16 Bt[64 * LDA];   // B tile transposed [n][k]
  A  += (long)blockIdx.z * sA;
  Bw += (long)blockIdx.z * sB;
  C  += (long)blockIdx.z * sC;
  const int tid = threadIdx.x, lane = tid & 31, wave = tid >> 5;
  const int l15 = lane & 15, lhalf = lane >> 4;
  const int wm = (wave >> 1) * 32, wn = (wave & 1) * 32;
  const int m0 = blockIdx.y * 64, n0 = blockIdx.x * 64;

  FFrag acc[2][2];
#pragma unroll
  for (int mi = 0; mi < 2; ++mi)
#pragma unroll
    for (int ni = 0; ni < 2; ++ni)
#pragma unroll
      for (int i = 0; i < 8; ++i) acc[mi][ni].f[i] = 0.f;

  for (int k0 = 0; k0 < K; k0 += 32) {
    // A tile 64x32: 512 float4 slots across 128 threads
#pragma unroll
    for (int p = 0; p < 4; ++p) {
      int idx = p * 128 + tid;
      int r = idx >> 3, c = (idx & 7) << 2;
      const float4 val = *(const float4*)(A + (long)(m0 + r) * K + (k0 + c));
      _Float16* d = &As[r * LDA + c];
      d[0] = (_Float16)val.x; d[1] = (_Float16)val.y;
      d[2] = (_Float16)val.z; d[3] = (_Float16)val.w;
    }
    // B tile 32x64 read coalesced along n, stored transposed [n][k]
#pragma unroll
    for (int p = 0; p < 4; ++p) {
      int idx = p * 128 + tid;
      int kr = idx >> 4, c = (idx & 15) << 2;
      const float4 val = *(const float4*)(Bw + (long)(k0 + kr) * N + (n0 + c));
      Bt[(c + 0) * LDA + kr] = (_Float16)val.x;
      Bt[(c + 1) * LDA + kr] = (_Float16)val.y;
      Bt[(c + 2) * LDA + kr] = (_Float16)val.z;
      Bt[(c + 3) * LDA + kr] = (_Float16)val.w;
    }
    __syncthreads();

    HFrag af[2], bf[2];
#pragma unroll
    for (int mi = 0; mi < 2; ++mi) {
      // A frag: M=lane&15, K interleaved {kb..kb+7, kb+16..kb+23}, kb=8*lhalf
      const _Float16* s = &As[(wm + mi * 16 + l15) * LDA + lhalf * 8];
      af[mi].h[0] = *(const v8h*)s;
      af[mi].h[1] = *(const v8h*)(s + 16);
    }
#pragma unroll
    for (int ni = 0; ni < 2; ++ni) {
      // B frag: N=lane&15, K interleaved {kb..kb+7, kb+16..kb+23}, kb=8*lhalf
      const _Float16* s = &Bt[(wn + ni * 16 + l15) * LDA + lhalf * 8];
      bf[ni].h[0] = *(const v8h*)s;
      bf[ni].h[1] = *(const v8h*)(s + 16);
    }
#pragma unroll
    for (int mi = 0; mi < 2; ++mi)
#pragma unroll
      for (int ni = 0; ni < 2; ++ni)
        acc[mi][ni].v = wmma_f16(af[mi], bf[ni], acc[mi][ni].v);
    __syncthreads();
  }

  // Branch-free epilogue: bias hoisted per column block; ACCUM loads batched.
#pragma unroll
  for (int ni = 0; ni < 2; ++ni) {
    const int gn = n0 + wn + ni * 16 + l15;           // N = lane&15
    float bv = 0.f;
    if (HAS_BIAS) bv = bias[gn];
#pragma unroll
    for (int mi = 0; mi < 2; ++mi) {
      const int gmb = m0 + wm + mi * 16 + lhalf * 8;  // C layout: M = i+8*lhalf
      float* cp = C + (long)gmb * N + gn;
      float prev[8];
      if (ACCUM) {
#pragma unroll
        for (int i = 0; i < 8; ++i) prev[i] = cp[(long)i * N];
      }
#pragma unroll
      for (int i = 0; i < 8; ++i) {
        float v = acc[mi][ni].f[i] + bv;
        if (ACCUM) v += prev[i];
        cp[(long)i * N] = v;
      }
    }
  }
}

// ---------------------------------------------------------------------------
// Fused multi-head attention (flash style): softmax(QK^T/8) V per (b,h).
// Grid: (N/64 q-blocks, B*H). 128 threads = 4 waves; each wave owns 16 q rows.
// Q/K/V layout (B,N,F) with head h at columns [h*64, h*64+64).
// The softmax row-SUM is computed on the matrix pipe (P x ones WMMA) instead
// of a ds_bpermute butterfly; only the row-max uses cross-lane shuffles.
// ---------------------------------------------------------------------------
__global__ __launch_bounds__(128) void attn_kernel(
    const float* __restrict__ Qf, const float* __restrict__ Kf,
    const float* __restrict__ Vf, float* __restrict__ Of)
{
  constexpr int LDK = 72, LDV = 40, LDP = 40;
  __shared__ __align__(16) _Float16 Ks[32 * LDK];      // [key][d]
  __shared__ __align__(16) _Float16 Vt[64 * LDV];      // [d][key] (transposed)
  __shared__ __align__(16) _Float16 Pw[4][16 * LDP];   // per-wave P buffer
  const int bh = blockIdx.y, b = bh >> 3, h = bh & 7;
  const int tid = threadIdx.x, lane = tid & 31, wave = tid >> 5;
  const int l15 = lane & 15, lhalf = lane >> 4, kb = lhalf * 8;
  const int qbase = blockIdx.x * 64 + wave * 16;
  const long hoff = (long)b * Nn * Fc + (long)h * Dc;

  // All-ones 32x16 B-matrix: D = P x ones -> per-row sums in C/D layout.
  HFrag onesB;
#pragma unroll
  for (int j = 0; j < 16; ++j) onesB.e[j] = (_Float16)1.0f;

  // Q fragments live in registers for the whole kernel (A layout, d-chunks of 32)
  HFrag qf[2];
  {
    const float* qs = Qf + hoff + (long)(qbase + l15) * Fc;
#pragma unroll
    for (int c = 0; c < 2; ++c)
#pragma unroll
      for (int j = 0; j < 8; ++j) {
        qf[c].e[j]     = (_Float16)qs[c * 32 + lhalf * 16 + j];
        qf[c].e[8 + j] = (_Float16)qs[c * 32 + lhalf * 16 + 8 + j];
      }
  }

  float m[8], l[8];
  FFrag o[4];
#pragma unroll
  for (int i = 0; i < 8; ++i) { m[i] = -1e30f; l[i] = 0.f; }
#pragma unroll
  for (int nb = 0; nb < 4; ++nb)
#pragma unroll
    for (int i = 0; i < 8; ++i) o[nb].f[i] = 0.f;

  for (int kv0 = 0; kv0 < Nn; kv0 += 32) {
    // Stage K (row-major) and V (transposed) tiles: 32 keys x 64 d, fp32->f16
#pragma unroll
    for (int p = 0; p < 4; ++p) {
      int idx = p * 128 + tid;
      int r = idx >> 4, c = (idx & 15) << 2;
      const float4 kv = *(const float4*)(Kf + hoff + (long)(kv0 + r) * Fc + c);
      _Float16* d = &Ks[r * LDK + c];
      d[0] = (_Float16)kv.x; d[1] = (_Float16)kv.y;
      d[2] = (_Float16)kv.z; d[3] = (_Float16)kv.w;
      const float4 vv = *(const float4*)(Vf + hoff + (long)(kv0 + r) * Fc + c);
      Vt[(c + 0) * LDV + r] = (_Float16)vv.x;
      Vt[(c + 1) * LDV + r] = (_Float16)vv.y;
      Vt[(c + 2) * LDV + r] = (_Float16)vv.z;
      Vt[(c + 3) * LDV + r] = (_Float16)vv.w;
    }
    __syncthreads();

    // S = Q K^T : two 16-key column blocks, K-dim d=64 chained over 2 WMMAs
    FFrag s2[2];
#pragma unroll
    for (int cb = 0; cb < 2; ++cb) {
      HFrag kf0, kf1;
      const _Float16* ks0 = &Ks[(cb * 16 + l15) * LDK + lhalf * 16];
      kf0.h[0] = *(const v8h*)ks0; kf0.h[1] = *(const v8h*)(ks0 + 8);
      const _Float16* ks1 = &Ks[(cb * 16 + l15) * LDK + 32 + lhalf * 16];
      kf1.h[0] = *(const v8h*)ks1; kf1.h[1] = *(const v8h*)(ks1 + 8);
      FFrag z;
#pragma unroll
      for (int i = 0; i < 8; ++i) z.f[i] = 0.f;
      v8f sv = wmma_f16(qf[0], kf0, z.v);
      sv = wmma_f16(qf[1], kf1, sv);
      s2[cb].v = sv;
    }

    // Online softmax: row-max butterfly over the 16-lane halves (rows i+8*lhalf)
    float sf[8];
#pragma unroll
    for (int i = 0; i < 8; ++i) {
      float a0 = s2[0].f[i] * 0.125f, a1 = s2[1].f[i] * 0.125f;
      s2[0].f[i] = a0; s2[1].f[i] = a1;
      float mx = fmaxf(a0, a1);
#pragma unroll
      for (int off = 1; off < 16; off <<= 1) mx = fmaxf(mx, __shfl_xor(mx, off));
      float mn = fmaxf(m[i], mx);
      sf[i] = __expf(m[i] - mn);
      m[i] = mn;
    }
    float p0[8], p1[8];
#pragma unroll
    for (int i = 0; i < 8; ++i) {
      p0[i] = __expf(s2[0].f[i] - m[i]);
      p1[i] = __expf(s2[1].f[i] - m[i]);
    }
#pragma unroll
    for (int nb = 0; nb < 4; ++nb)
#pragma unroll
      for (int i = 0; i < 8; ++i) o[nb].f[i] *= sf[i];

    // P (C layout) -> per-wave LDS -> reload in A-fragment layout
    _Float16* pl = Pw[wave];
#pragma unroll
    for (int i = 0; i < 8; ++i) {
      int row = i + lhalf * 8;
      pl[row * LDP + l15]      = (_Float16)p0[i];
      pl[row * LDP + 16 + l15] = (_Float16)p1[i];
    }
    HFrag pa;
    const _Float16* ps = &pl[l15 * LDP + lhalf * 16];
    pa.h[0] = *(const v8h*)ps;
    pa.h[1] = *(const v8h*)(ps + 8);

    // Row sums on the matrix pipe: lw[i] = sum_k P[row i][k] (bcast over lanes)
    {
      FFrag z;
#pragma unroll
      for (int i = 0; i < 8; ++i) z.f[i] = 0.f;
      FFrag lw; lw.v = wmma_f16(pa, onesB, z.v);
#pragma unroll
      for (int i = 0; i < 8; ++i) l[i] = l[i] * sf[i] + lw.f[i];
    }

    // O += P V : 4 d-column blocks of 16
#pragma unroll
    for (int nb = 0; nb < 4; ++nb) {
      HFrag vf;
      const _Float16* vs = &Vt[(nb * 16 + l15) * LDV + lhalf * 16];
      vf.h[0] = *(const v8h*)vs; vf.h[1] = *(const v8h*)(vs + 8);
      o[nb].v = wmma_f16(pa, vf, o[nb].v);
    }
    __syncthreads();
  }

#pragma unroll
  for (int i = 0; i < 8; ++i) {
    float r = 1.f / l[i];
    int q = qbase + i + lhalf * 8;
#pragma unroll
    for (int nb = 0; nb < 4; ++nb)
      Of[hoff + (long)q * Fc + nb * 16 + l15] = o[nb].f[i] * r;
  }
}

// ---------------------------------------------------------------------------
// Geometry scores: GS[b,q,k] = (gq.gk)/8 + mean_h(distMLP(dist(q,k)))
// WMMA over K=512 + fused distance-MLP epilogue (mean over heads folded into
// d2m[j] = mean_h d2_w[j][h]).  Grid: (N/64 k-blocks, N/64 q-blocks, B).
// ---------------------------------------------------------------------------
__global__ __launch_bounds__(128) void gscore_kernel(
    const float* __restrict__ GQ, const float* __restrict__ GK,
    const float* __restrict__ coords,
    const float* __restrict__ d1w, const float* __restrict__ d1b,
    const float* __restrict__ d2w, const float* __restrict__ d2b,
    float* __restrict__ GS)
{
  constexpr int LDA = 40;
  __shared__ __align__(16) _Float16 As[64 * LDA];   // gq rows [q][d]
  __shared__ __align__(16) _Float16 Bs[64 * LDA];   // gk rows [key][d]
  const int b = blockIdx.z;
  const float* Aq = GQ + (long)b * Nn * Fc;
  const float* Ak = GK + (long)b * Nn * Fc;
  const int tid = threadIdx.x, lane = tid & 31, wave = tid >> 5;
  const int l15 = lane & 15, lhalf = lane >> 4;
  const int wm = (wave >> 1) * 32, wn = (wave & 1) * 32;
  const int m0 = blockIdx.y * 64, n0 = blockIdx.x * 64;

  FFrag acc[2][2];
#pragma unroll
  for (int mi = 0; mi < 2; ++mi)
#pragma unroll
    for (int ni = 0; ni < 2; ++ni)
#pragma unroll
      for (int i = 0; i < 8; ++i) acc[mi][ni].f[i] = 0.f;

  for (int k0 = 0; k0 < Fc; k0 += 32) {
#pragma unroll
    for (int p = 0; p < 4; ++p) {
      int idx = p * 128 + tid;
      int r = idx >> 3, c = (idx & 7) << 2;
      const float4 qa = *(const float4*)(Aq + (long)(m0 + r) * Fc + (k0 + c));
      _Float16* da = &As[r * LDA + c];
      da[0] = (_Float16)qa.x; da[1] = (_Float16)qa.y;
      da[2] = (_Float16)qa.z; da[3] = (_Float16)qa.w;
      const float4 ka = *(const float4*)(Ak + (long)(n0 + r) * Fc + (k0 + c));
      _Float16* db = &Bs[r * LDA + c];
      db[0] = (_Float16)ka.x; db[1] = (_Float16)ka.y;
      db[2] = (_Float16)ka.z; db[3] = (_Float16)ka.w;
    }
    __syncthreads();

    HFrag af[2], bf[2];
#pragma unroll
    for (int mi = 0; mi < 2; ++mi) {
      const _Float16* s = &As[(wm + mi * 16 + l15) * LDA + lhalf * 8];
      af[mi].h[0] = *(const v8h*)s;
      af[mi].h[1] = *(const v8h*)(s + 16);
    }
#pragma unroll
    for (int ni = 0; ni < 2; ++ni) {
      // B[k=d][n=key] = gk[key][d]: row-major gk rows give contiguous K reads
      const _Float16* s = &Bs[(wn + ni * 16 + l15) * LDA + lhalf * 8];
      bf[ni].h[0] = *(const v8h*)s;
      bf[ni].h[1] = *(const v8h*)(s + 16);
    }
#pragma unroll
    for (int mi = 0; mi < 2; ++mi)
#pragma unroll
      for (int ni = 0; ni < 2; ++ni)
        acc[mi][ni].v = wmma_f16(af[mi], bf[ni], acc[mi][ni].v);
    __syncthreads();
  }

  // Distance-MLP constants (tiny, L2-cached)
  float w1[8], bb1[8], d2m[8], b2m = 0.f;
#pragma unroll
  for (int j = 0; j < 8; ++j) { w1[j] = d1w[j]; bb1[j] = d1b[j]; }
#pragma unroll
  for (int j = 0; j < 8; ++j) {
    float s = 0.f;
#pragma unroll
    for (int hh = 0; hh < 8; ++hh) s += d2w[j * 8 + hh];
    d2m[j] = s * 0.125f;
  }
#pragma unroll
  for (int hh = 0; hh < 8; ++hh) b2m += d2b[hh];
  b2m *= 0.125f;

  // Hoisted coords: key coords depend only on ni (2/lane), query on (mi,i).
  float ckx[2], cky[2], ckz[2];
#pragma unroll
  for (int ni = 0; ni < 2; ++ni) {
    const int k = n0 + wn + ni * 16 + l15;
    const float* ck = coords + ((long)b * Nn + k) * 3;
    ckx[ni] = ck[0]; cky[ni] = ck[1]; ckz[ni] = ck[2];
  }

#pragma unroll
  for (int mi = 0; mi < 2; ++mi)
#pragma unroll
    for (int i = 0; i < 8; ++i) {
      const int q = m0 + wm + mi * 16 + i + lhalf * 8;
      const float* cq = coords + ((long)b * Nn + q) * 3;
      const float cqx = cq[0], cqy = cq[1], cqz = cq[2];
#pragma unroll
      for (int ni = 0; ni < 2; ++ni) {
        const int k = n0 + wn + ni * 16 + l15;
        float dx = cqx - ckx[ni], dy = cqy - cky[ni], dz = cqz - ckz[ni];
        float dist = sqrtf(fmaxf(dx * dx + dy * dy + dz * dz, 1e-12f));
        float dt = b2m;
#pragma unroll
        for (int j = 0; j < 8; ++j)
          dt += gelu_tanh(dist * w1[j] + bb1[j]) * d2m[j];
        GS[((long)b * Nn + q) * Nn + k] = acc[mi][ni].f[i] * 0.125f + dt;
      }
    }
}

// ---------------------------------------------------------------------------
// Row softmax over 1024-wide rows, in place. One 128-thread block per row.
// ---------------------------------------------------------------------------
__global__ __launch_bounds__(128) void softmax_row_kernel(float* __restrict__ GS)
{
  float* p = GS + (long)blockIdx.x * Nn;
  const int tid = threadIdx.x, lane = tid & 31, wave = tid >> 5;
  __shared__ float rmax[4], rsum[4];
  float v[8];
#pragma unroll
  for (int j = 0; j < 8; ++j) v[j] = p[tid + j * 128];
  float mx = v[0];
#pragma unroll
  for (int j = 1; j < 8; ++j) mx = fmaxf(mx, v[j]);
#pragma unroll
  for (int off = 1; off < 32; off <<= 1) mx = fmaxf(mx, __shfl_xor(mx, off));
  if (lane == 0) rmax[wave] = mx;
  __syncthreads();
  mx = fmaxf(fmaxf(rmax[0], rmax[1]), fmaxf(rmax[2], rmax[3]));
  float s = 0.f;
#pragma unroll
  for (int j = 0; j < 8; ++j) { v[j] = __expf(v[j] - mx); s += v[j]; }
#pragma unroll
  for (int off = 1; off < 32; off <<= 1) s += __shfl_xor(s, off);
  if (lane == 0) rsum[wave] = s;
  __syncthreads();
  float r = 1.f / (rsum[0] + rsum[1] + rsum[2] + rsum[3]);
#pragma unroll
  for (int j = 0; j < 8; ++j) p[tid + j * 128] = v[j] * r;
}

// ---------------------------------------------------------------------------
// Host orchestration. Workspace layout (floats):
//   Q,K,V,ATT,SUM,GQ,GK,GV : 8 buffers of B*N*F = 4,194,304 floats each
//   GS : B*N*N = 8,388,608 floats
// ---------------------------------------------------------------------------
extern "C" void kernel_launch(void* const* d_in, const int* in_sizes, int n_in,
                              void* d_out, int out_size, void* d_ws, size_t ws_size,
                              hipStream_t stream) {
  (void)in_sizes; (void)n_in; (void)out_size; (void)ws_size;
  const float* features = (const float*)d_in[0];
  const float* geometry = (const float*)d_in[1];
  const float* coords   = (const float*)d_in[2];
  const float* wq = (const float*)d_in[3];   const float* bq = (const float*)d_in[4];
  const float* wk = (const float*)d_in[5];   const float* bk = (const float*)d_in[6];
  const float* wv = (const float*)d_in[7];   const float* bv = (const float*)d_in[8];
  const float* wo = (const float*)d_in[9];   const float* bo = (const float*)d_in[10];
  const float* gq_w = (const float*)d_in[11]; const float* gq_b = (const float*)d_in[12];
  const float* gk_w = (const float*)d_in[13]; const float* gk_b = (const float*)d_in[14];
  const float* gv_w = (const float*)d_in[15]; const float* gv_b = (const float*)d_in[16];
  const float* d1w = (const float*)d_in[17];  const float* d1b = (const float*)d_in[18];
  const float* d2w = (const float*)d_in[19];  const float* d2b = (const float*)d_in[20];
  const float* opw = (const float*)d_in[21];  const float* opb = (const float*)d_in[22];
  float* out = (float*)d_out;

  const long BNF = (long)Bc * Nn * Fc;
  float* ws  = (float*)d_ws;
  float* Q   = ws + 0 * BNF;
  float* Kb  = ws + 1 * BNF;
  float* Vb  = ws + 2 * BNF;
  float* ATT = ws + 3 * BNF;
  float* SUM = ws + 4 * BNF;
  float* GQ  = ws + 5 * BNF;
  float* GK  = ws + 6 * BNF;
  float* GV  = ws + 7 * BNF;
  float* GS  = ws + 8 * BNF;

  const int M = Bc * Nn;             // 8192
  dim3 blk(128);
  dim3 gProj(Fc / 64, M / 64, 1);    // (8,128)

  // Feature branch: q/k/v projections -> fused attention -> wo
  gemm_wmma_kernel<true, false><<<gProj, blk, 0, stream>>>(features, wq, bq, Q,  M, Fc, Fc, 0, 0, 0);
  gemm_wmma_kernel<true, false><<<gProj, blk, 0, stream>>>(features, wk, bk, Kb, M, Fc, Fc, 0, 0, 0);
  gemm_wmma_kernel<true, false><<<gProj, blk, 0, stream>>>(features, wv, bv, Vb, M, Fc, Fc, 0, 0, 0);
  attn_kernel<<<dim3(Nn / 64, Bc * Hc), blk, 0, stream>>>(Q, Kb, Vb, ATT);
  gemm_wmma_kernel<true, false><<<gProj, blk, 0, stream>>>(ATT, wo, bo, SUM, M, Fc, Fc, 0, 0, 0);

  // Geometry branch: gq/gk/gv -> scores(+dist MLP) -> softmax -> attn@gv (+= SUM)
  gemm_wmma_kernel<true, false><<<gProj, blk, 0, stream>>>(geometry, gq_w, gq_b, GQ, M, Gc, Fc, 0, 0, 0);
  gemm_wmma_kernel<true, false><<<gProj, blk, 0, stream>>>(geometry, gk_w, gk_b, GK, M, Gc, Fc, 0, 0, 0);
  gemm_wmma_kernel<true, false><<<gProj, blk, 0, stream>>>(geometry, gv_w, gv_b, GV, M, Gc, Fc, 0, 0, 0);
  gscore_kernel<<<dim3(Nn / 64, Nn / 64, Bc), blk, 0, stream>>>(
      GQ, GK, coords, d1w, d1b, d2w, d2b, GS);
  softmax_row_kernel<<<dim3(Bc * Nn), blk, 0, stream>>>(GS);
  gemm_wmma_kernel<false, true><<<dim3(Fc / 64, Nn / 64, Bc), blk, 0, stream>>>(
      GS, GV, nullptr, SUM, Nn, Nn, Fc,
      (long)Nn * Nn, (long)Nn * Fc, (long)Nn * Fc);

  // Final projection: out = SUM @ op_w + op_b
  gemm_wmma_kernel<true, false><<<gProj, blk, 0, stream>>>(SUM, opw, opb, out, M, Fc, Fc, 0, 0, 0);
}
